// SpatialRegularizationLayer_64149631533715
// MI455X (gfx1250) — matrix operand order repaired and checked by reading, not compiled, not run on hardware
//
#include <hip/hip_runtime.h>

// ---------------------------------------------------------------------------
// Fused SpatialRegularizationLayer for MI455X (gfx1250, wave32, WMMA bf16).
//   conv1(1x1,80->512)+b1 -> conv2(1x1,512->512)+b2 -> grouped 14x14 dot
//   (512 groups x 4) -> relu -> fc(2048->80)+fcb
// One workgroup per batch element; h1 lives entirely in LDS (255KB < 320KB
// WGP LDS); h2 is consumed in registers by the conv3 epilogue.
// x is pre-converted to bf16 (padded/transposed) and staged into LDS with
// GLOBAL_LOAD_ASYNC_TO_LDS_B128 (ASYNCcnt) when available.
// ---------------------------------------------------------------------------

typedef __attribute__((ext_vector_type(16))) __bf16 v16bf;
typedef __attribute__((ext_vector_type(8)))  float  v8f;
typedef int v4i_ __attribute__((vector_size(16)));   // raw vector for async-LDS builtin

union Frag16 { v16bf v; uint4 q[2]; };

#define C_IN   80     // input channels
#define KP1    96     // conv1 K padded to 3 x 32
#define HW     196    // 14*14 spatial
#define NP     208    // pixels padded to 13 x 16
#define C1     512    // hidden channels / groups
#define NT1    13     // n tiles
#define MT1    32     // m tiles (512/16)
#define SMEM_BYTES (NP*KP1*2 + NP*C1*2 + 2048*4)   // 39936 + 212992 + 8192
#define N_W1   (C1*KP1)          // bf16 elements of padded w1
#define N_W2   (C1*C1)           // bf16 elements of w2
#define N_XB   (256*NP*KP1)      // bf16 elements of padded/transposed x

#if defined(__AMDGCN__) && __has_builtin(__builtin_amdgcn_global_load_async_to_lds_b128)
#define HAVE_ASYNC_LDS 1
#else
#define HAVE_ASYNC_LDS 0
#endif

// f32 -> bf16 (RNE). Native fptrunc: backend picks hardware cvt if available.
__device__ __forceinline__ unsigned short f2bf(float f) {
  return __builtin_bit_cast(unsigned short, (__bf16)f);
}
__device__ __forceinline__ unsigned pack2bf(float lo, float hi) {
  return (unsigned)f2bf(lo) | ((unsigned)f2bf(hi) << 16);
}

// ---------------- weight/x prep (fp32 -> bf16, pad, transpose) -------------
__global__ void prep_convert(const float* __restrict__ w1,
                             const float* __restrict__ w2,
                             const float* __restrict__ x,
                             unsigned short* __restrict__ wsbf,
                             int do_x) {
  int idx = blockIdx.x * blockDim.x + threadIdx.x;
  if (idx < N_W1) {                       // padded w1: [512][96]
    int r = idx / KP1, c = idx % KP1;
    float v = (c < C_IN) ? w1[r * C_IN + c] : 0.f;
    wsbf[idx] = f2bf(v);
  } else if (idx < N_W1 + N_W2) {         // w2: [512][512]
    wsbf[idx] = f2bf(w2[idx - N_W1]);
  } else if (do_x && idx < N_W1 + N_W2 + N_XB) {
    // x: [B][80][196] f32 -> [B][208 pixels][96 ch] bf16 (pixel-major, padded)
    int j = idx - (N_W1 + N_W2);
    int b = j / (NP * KP1);
    int r = j % (NP * KP1);
    int p = r / KP1, c = r % KP1;
    float v = (p < HW && c < C_IN) ? x[(size_t)b * C_IN * HW + c * HW + p] : 0.f;
    wsbf[idx] = f2bf(v);
  }
}

// ------------------------------ fused kernel -------------------------------
__global__ __launch_bounds__(256, 1)
void srl_fused(const float* __restrict__ x,
               const float* __restrict__ b1,
               const float* __restrict__ b2,
               const float* __restrict__ w3,
               const float* __restrict__ fcw,
               const float* __restrict__ fcb,
               const unsigned short* __restrict__ w1bf,   // [512][96] bf16
               const unsigned short* __restrict__ w2bf,   // [512][512] bf16
               const unsigned short* __restrict__ xbf,    // [B][208][96] bf16
               int use_pre,
               float* __restrict__ out) {
  extern __shared__ __align__(16) char smem[];
  unsigned short* xs  = (unsigned short*)smem;                        // [NP][KP1]
  unsigned short* h1s = (unsigned short*)(smem + NP*KP1*2);           // [NP][C1]
  float*          gs  = (float*)(smem + NP*KP1*2 + NP*C1*2);          // [2048]

  const int b     = blockIdx.x;
  const int tid   = threadIdx.x;
  const int wave  = tid >> 5;
  const int lane  = tid & 31;
  const int col   = lane & 15;      // N column / A row within tile
  const int khalf = lane >> 4;      // 0: lanes 0-15, 1: lanes 16-31

  // ---- Phase 1: stage x[b] tile (bf16, pixel-major, padded) into LDS ------
  if (use_pre) {
    const unsigned short* xb = xbf + (size_t)b * NP * KP1;
#if HAVE_ASYNC_LDS
    // Async DMA into LDS: 16B per lane per op, tracked by ASYNCcnt.
    __attribute__((address_space(1))) v4i_* g =
        (__attribute__((address_space(1))) v4i_*)xb;
    __attribute__((address_space(3))) v4i_* l =
        (__attribute__((address_space(3))) v4i_*)xs;
    for (int c = tid; c < (NP * KP1) / 8; c += 256)
      __builtin_amdgcn_global_load_async_to_lds_b128(g + c, l + c, 0, 0);
#if __has_builtin(__builtin_amdgcn_s_wait_asynccnt)
    __builtin_amdgcn_s_wait_asynccnt(0);
#else
    asm volatile("s_wait_asynccnt 0" ::: "memory");
#endif
#else
    for (int c = tid; c < (NP * KP1) / 8; c += 256)
      ((uint4*)xs)[c] = ((const uint4*)xb)[c];
#endif
  } else {
    // Fallback: convert fp32 x in-kernel (workspace too small for bf16 x).
    const float* xb = x + (size_t)b * C_IN * HW;
    for (int idx = tid; idx < NP * KP1; idx += 256) {
      int p = idx / KP1, c = idx % KP1;
      float v = (p < HW && c < C_IN) ? xb[c * HW + p] : 0.f;
      xs[idx] = f2bf(v);
    }
  }
  __syncthreads();

  // ---- Phase 2: GEMM1  h1[512 x 196] = w1 @ x + b1  (bf16 WMMA) ----------
  // Each wave owns 4 m-tiles; A fragments + bias hoisted, reused over 13 nt.
  for (int mi = 0; mi < 4; ++mi) {
    int mt = wave + mi * 8;
    int m0 = mt*16 + khalf*8;
    float bias[8];
    for (int i = 0; i < 8; ++i) bias[i] = b1[m0 + i];
    Frag16 a[KP1 / 32];
    for (int kk = 0; kk < KP1 / 32; ++kk) {
      // A: row = mt*16+col, two 16B chunks at K = kk*32 + khalf*8 (+16)
      const uint4* ap = (const uint4*)(w1bf + (mt*16 + col) * KP1 + kk*32 + khalf*8);
      a[kk].q[0] = ap[0]; a[kk].q[1] = ap[2];
    }
    for (int nt = 0; nt < NT1; ++nt) {
      v8f acc = {0.f,0.f,0.f,0.f,0.f,0.f,0.f,0.f};
      for (int kk = 0; kk < KP1 / 32; ++kk) {
        Frag16 bm;
        // B: column n = nt*16+col, two 16B chunks at K = kk*32 + khalf*8 (+16)
        const uint4* bp = (const uint4*)(xs + (nt*16 + col) * KP1 + kk*32 + khalf*8);
        bm.q[0] = bp[0]; bm.q[1] = bp[2];
        acc = __builtin_amdgcn_wmma_f32_16x16x32_bf16(false, a[kk].v, false, bm.v,
                                                      (short)0, acc, false, false);
      }
      // epilogue: + bias, convert to bf16, store to h1s[pixel][channel]
      int n = nt*16 + col;
      uint4 st;
      st.x = pack2bf(acc[0] + bias[0], acc[1] + bias[1]);
      st.y = pack2bf(acc[2] + bias[2], acc[3] + bias[3]);
      st.z = pack2bf(acc[4] + bias[4], acc[5] + bias[5]);
      st.w = pack2bf(acc[6] + bias[6], acc[7] + bias[7]);
      *(uint4*)(h1s + n * C1 + m0) = st;
    }
  }
  __syncthreads();

  // ---- Phase 3: GEMM2 (w2 @ h1 + b2) fused with conv3 spatial reduction --
  for (int mi = 0; mi < 4; ++mi) {
    int mt = wave + mi * 8;           // this wave's output-channel tile
    int g0 = mt*16 + khalf*8;         // first of this lane's 8 groups
    const unsigned short* w2row = w2bf + (mt*16 + col) * C1;
    float bias2[8];
    for (int i = 0; i < 8; ++i) bias2[i] = b2[g0 + i];
    v8f accs[NT1];
    for (int nt = 0; nt < NT1; ++nt)
      accs[nt] = (v8f){0.f,0.f,0.f,0.f,0.f,0.f,0.f,0.f};

    for (int kk = 0; kk < C1 / 32; ++kk) {
      Frag16 a;
      const uint4* ap = (const uint4*)(w2row + kk*32 + khalf*8);
      a.q[0] = ap[0]; a.q[1] = ap[2];            // A reused for all 13 n-tiles
      if (kk + 1 < C1 / 32)
        __builtin_prefetch(w2row + (kk + 1) * 32, 0, 2);   // global_prefetch_b8
      for (int nt = 0; nt < NT1; ++nt) {
        Frag16 bm;
        const uint4* bp = (const uint4*)(h1s + (nt*16 + col) * C1 + kk*32 + khalf*8);
        bm.q[0] = bp[0]; bm.q[1] = bp[2];
        accs[nt] = __builtin_amdgcn_wmma_f32_16x16x32_bf16(false, a.v, false, bm.v,
                                                           (short)0, accs[nt], false, false);
      }
    }

    // conv3: g[g,o] += h2[g,p] * w3[g,o,p], p = this lane's pixel column
    float gacc[8][4] = {{0.f}};
    for (int nt = 0; nt < NT1; ++nt) {
      int p = nt*16 + col;
      if (p < HW) {
        for (int i = 0; i < 8; ++i) {
          float hv = accs[nt][i] + bias2[i];
          const float* wp = w3 + (size_t)(g0 + i) * (4*HW) + p;
          gacc[i][0] += hv * wp[0];
          gacc[i][1] += hv * wp[HW];
          gacc[i][2] += hv * wp[2*HW];
          gacc[i][3] += hv * wp[3*HW];
        }
      }
    }
    // reduce the 16 pixel columns inside each wave half, relu, write LDS
    for (int i = 0; i < 8; ++i) {
      for (int o = 0; o < 4; ++o) {
        float v = gacc[i][o];
        v += __shfl_xor(v, 1, 32);
        v += __shfl_xor(v, 2, 32);
        v += __shfl_xor(v, 4, 32);
        v += __shfl_xor(v, 8, 32);
        if (col == 0) gs[(g0 + i) * 4 + o] = v > 0.f ? v : 0.f;
      }
    }
  }
  __syncthreads();

  // ---- Phase 4: FC  out[b,l] = fc_w[l,:] . g + fc_b[l] --------------------
  for (int li = 0; li < 10; ++li) {
    int l = wave * 10 + li;           // 8 waves x 10 = 80 outputs
    float s = 0.f;
    const float* wr = fcw + (size_t)l * 2048;
    for (int j = lane; j < 2048; j += 32) s += wr[j] * gs[j];
    s += __shfl_xor(s, 16, 32);
    s += __shfl_xor(s,  8, 32);
    s += __shfl_xor(s,  4, 32);
    s += __shfl_xor(s,  2, 32);
    s += __shfl_xor(s,  1, 32);
    if (lane == 0) out[b * 80 + l] = s + fcb[l];
  }
}

// ------------------------------- launcher ----------------------------------
extern "C" void kernel_launch(void* const* d_in, const int* in_sizes, int n_in,
                              void* d_out, int out_size, void* d_ws, size_t ws_size,
                              hipStream_t stream) {
  (void)in_sizes; (void)n_in; (void)out_size;
  const float* x   = (const float*)d_in[0];
  const float* w1  = (const float*)d_in[1];
  const float* b1  = (const float*)d_in[2];
  const float* w2  = (const float*)d_in[3];
  const float* b2  = (const float*)d_in[4];
  const float* w3  = (const float*)d_in[5];
  const float* fcw = (const float*)d_in[6];
  const float* fcb = (const float*)d_in[7];
  float* out = (float*)d_out;

  unsigned short* wsbf = (unsigned short*)d_ws;
  const size_t need_pre = (size_t)(N_W1 + N_W2 + N_XB) * 2;
  const int use_pre = (ws_size >= need_pre) ? 1 : 0;

  const int total = use_pre ? (N_W1 + N_W2 + N_XB) : (N_W1 + N_W2);
  prep_convert<<<(total + 255) / 256, 256, 0, stream>>>(w1, w2, x, wsbf, use_pre);

  (void)hipFuncSetAttribute((const void*)srl_fused,
                            hipFuncAttributeMaxDynamicSharedMemorySize, SMEM_BYTES);
  srl_fused<<<256, 256, SMEM_BYTES, stream>>>(x, b1, b2, w3, fcw, fcb,
                                              wsbf,                 // w1bf
                                              wsbf + N_W1,          // w2bf
                                              wsbf + N_W1 + N_W2,   // xbf
                                              use_pre, out);
}
